// LinearTemporalDiffusionTransformerLayer_2_31379031064720
// MI455X (gfx1250) — compile-verified, matrix operand order checked by reading, not run on hardware
//
#include <hip/hip_runtime.h>

// ============================================================================
// LinearTemporalDiffusionTransformerLayer on gfx1250 (MI455X), bf16 WMMA path.
// Heavy GEMMs: v_wmma_f32_16x16x32_bf16, fp32 accumulation, 128x128 block
// tiles, 32x64 wave tiles (2x4 WMMA), async global->LDS staging via the
// gfx1250 GLOBAL_LOAD_ASYNC_TO_LDS_B128 builtin (typed v4i addrspace ptrs).
// ============================================================================

typedef __attribute__((ext_vector_type(16))) __bf16 v16bf;
typedef __attribute__((ext_vector_type(8)))  float  v8f;
typedef int v4i_ __attribute__((vector_size(16)));

// ---- gfx1250 async-to-LDS path (probe via __has_builtin; safe fallback) ----
#if defined(__has_builtin)
#  if __has_builtin(__builtin_amdgcn_global_load_async_to_lds_b128)
#    define USE_ASYNC_LDS 1
#  endif
#  if __has_builtin(__builtin_amdgcn_s_wait_asynccnt)
#    define WAIT_ASYNC() __builtin_amdgcn_s_wait_asynccnt(0)
#  endif
#endif
#ifndef USE_ASYNC_LDS
#  define USE_ASYNC_LDS 0
#endif
#ifndef WAIT_ASYNC
#  define WAIT_ASYNC() asm volatile("s_wait_asynccnt 0" ::: "memory")
#endif

// builtin wants v4i pointers: addrspace(1) for global, addrspace(3) for LDS
#define ASG(p) ((__attribute__((address_space(1))) v4i_*)(p))
#define ASL(p) ((__attribute__((address_space(3))) v4i_*)(p))

__device__ __forceinline__ unsigned short f2bf(float f) {
    unsigned int u = __float_as_uint(f);
    unsigned int lsb = (u >> 16) & 1u;
    u += 0x7FFFu + lsb;               // round-to-nearest-even
    return (unsigned short)(u >> 16);
}

// ---------------------------------------------------------------------------
// Elementwise kernels (bandwidth-bound; trivial at 23.3 TB/s)
// ---------------------------------------------------------------------------
__global__ void f2bf_kernel(const float* __restrict__ in, unsigned short* __restrict__ out, long long n) {
    long long i = (long long)blockIdx.x * blockDim.x + threadIdx.x;
    long long stride = (long long)gridDim.x * blockDim.x;
    for (; i < n; i += stride) out[i] = f2bf(in[i]);
}

__global__ void silu_bf16_kernel(const float* __restrict__ in, unsigned short* __restrict__ out, long long n) {
    long long i = (long long)blockIdx.x * blockDim.x + threadIdx.x;
    long long stride = (long long)gridDim.x * blockDim.x;
    for (; i < n; i += stride) {
        float x = in[i];
        out[i] = f2bf(x / (1.0f + __expf(-x)));
    }
}

__global__ void gelu_bf16_kernel(const float* __restrict__ in, unsigned short* __restrict__ out, long long n) {
    long long i = (long long)blockIdx.x * blockDim.x + threadIdx.x;
    long long stride = (long long)gridDim.x * blockDim.x;
    for (; i < n; i += stride) {
        float x = in[i];
        out[i] = f2bf(0.5f * x * (1.0f + erff(x * 0.70710678118654752f)));
    }
}

__global__ void maskmul_bf16_kernel(const float* __restrict__ v, const float* __restrict__ mask,
                                    unsigned short* __restrict__ out, long long n, int D) {
    long long i = (long long)blockIdx.x * blockDim.x + threadIdx.x;
    long long stride = (long long)gridDim.x * blockDim.x;
    for (; i < n; i += stride) out[i] = f2bf(v[i] * mask[i / D]);
}

// ---------------------------------------------------------------------------
// Block reduction helper (wave32 shuffles + LDS across 8 waves)
// ---------------------------------------------------------------------------
__device__ __forceinline__ float block_sum(float v, float* sh) {
    for (int off = 16; off; off >>= 1) v += __shfl_xor(v, off, 32);
    int wid = threadIdx.x >> 5, lane = threadIdx.x & 31;
    if (lane == 0) sh[wid] = v;
    __syncthreads();
    int nw = blockDim.x >> 5;
    float r = (lane < nw) ? sh[lane] : 0.0f;
    for (int off = 16; off; off >>= 1) r += __shfl_xor(r, off, 32);
    __syncthreads();
    return r;
}

__global__ __launch_bounds__(256)
void ln_bf16_kernel(const float* __restrict__ x, const float* __restrict__ g,
                    const float* __restrict__ b, unsigned short* __restrict__ out, int C) {
    __shared__ float sh[8];
    long long row = blockIdx.x;
    const float* xr = x + row * C;
    float s = 0.f, s2 = 0.f;
    for (int i = threadIdx.x; i < C; i += blockDim.x) { float v = xr[i]; s += v; s2 += v * v; }
    float mean = block_sum(s, sh) / C;
    float var  = block_sum(s2, sh) / C - mean * mean;
    float inv  = rsqrtf(var + 1e-5f);
    for (int i = threadIdx.x; i < C; i += blockDim.x)
        out[row * C + i] = f2bf((xr[i] - mean) * inv * g[i] + b[i]);
}

// StylizationBlock core: out = silu( LN(y)*(1+scale) + shift ) as bf16.
__global__ __launch_bounds__(256)
void styl_silu_bf16_kernel(const float* __restrict__ y, const float* __restrict__ e,
                           const float* __restrict__ g, const float* __restrict__ b,
                           unsigned short* __restrict__ out, int C, int Tlen) {
    __shared__ float sh[8];
    long long row = blockIdx.x;
    long long bidx = row / Tlen;
    const float* yr = y + row * C;
    const float* er = e + bidx * 2 * C;
    float s = 0.f, s2 = 0.f;
    for (int i = threadIdx.x; i < C; i += blockDim.x) { float v = yr[i]; s += v; s2 += v * v; }
    float mean = block_sum(s, sh) / C;
    float var  = block_sum(s2, sh) / C - mean * mean;
    float inv  = rsqrtf(var + 1e-5f);
    for (int i = threadIdx.x; i < C; i += blockDim.x) {
        float h = (yr[i] - mean) * inv * g[i] + b[i];
        h = h * (1.0f + er[i]) + er[C + i];
        out[row * C + i] = f2bf(h / (1.0f + __expf(-h)));
    }
}

// Per-head softmax over Dh=64 (last axis). One wave per head; 16 waves/block.
__global__ __launch_bounds__(512)
void head_softmax_bf16_kernel(const float* __restrict__ q, unsigned short* __restrict__ out) {
    long long row = blockIdx.x;
    int wave = threadIdx.x >> 5;
    int lane = threadIdx.x & 31;
    long long base = row * 1024 + wave * 64;
    float a = q[base + lane], b = q[base + 32 + lane];
    float m = fmaxf(a, b);
    for (int off = 16; off; off >>= 1) m = fmaxf(m, __shfl_xor(m, off, 32));
    float ea = __expf(a - m), eb = __expf(b - m);
    float s = ea + eb;
    for (int off = 16; off; off >>= 1) s += __shfl_xor(s, off, 32);
    float inv = 1.0f / s;
    out[base + lane]      = f2bf(ea * inv);
    out[base + 32 + lane] = f2bf(eb * inv);
}

// Column softmax over axis=1 (length Tlen) per (b,d), optional mask, writing
// TRANSPOSED bf16 kT so the batched WMMA GEMM reads K contiguously.
__global__ __launch_bounds__(256)
void colsoftmax_T_kernel(const float* __restrict__ k, const float* __restrict__ mask,
                         unsigned short* __restrict__ outT, int Tlen, int D) {
    int b = blockIdx.y;
    int d = blockIdx.x * blockDim.x + threadIdx.x;
    const float* base = k + ((long long)b * Tlen) * D + d;
    const float* mb = mask ? (mask + (long long)b * Tlen) : (const float*)0;
    float m = -3.0e38f;
    for (int t = 0; t < Tlen; ++t) {
        float v = base[(long long)t * D];
        if (mb) v += (1.0f - mb[t]) * -1.0e6f;
        m = fmaxf(m, v);
    }
    float s = 0.f;
    for (int t = 0; t < Tlen; ++t) {
        float v = base[(long long)t * D];
        if (mb) v += (1.0f - mb[t]) * -1.0e6f;
        s += __expf(v - m);
    }
    float inv = 1.0f / s;
    int h = d >> 6, dd = d & 63;
    unsigned short* o = outT + (((long long)b * 16 + h) * 64 + dd) * Tlen;
    for (int t = 0; t < Tlen; ++t) {
        float v = base[(long long)t * D];
        if (mb) v += (1.0f - mb[t]) * -1.0e6f;
        o[t] = f2bf(__expf(v - m) * inv);
    }
}

// ---------------------------------------------------------------------------
// Tiled bf16 WMMA GEMM: C = A(MxK) * B(KxN) [+bias] [+res]
// Block = 256 threads = 8 waves; block tile 128x128; K step 32.
// Wave tile 32x64 = 2x4 v_wmma_f32_16x16x32_bf16 per K step (8 WMMA / 6 frags).
// A tile staged via gfx1250 GLOBAL_LOAD_ASYNC_TO_LDS_B128 when available.
// EPI: 0 = f32 out, 1 = f32 out + residual add, 2 = bf16 out.
// Requires K % 32 == 0, N % 8 == 0; M, N otherwise arbitrary (guarded).
// ---------------------------------------------------------------------------
#define TILE_M 128
#define TILE_N 128
#define TILE_K 32
#define LDSA_S 40   // A LDS row stride (elements); 80 B = multiple of 16
#define LDSB_S 34   // BT LDS row stride (elements); even for u32 access

union FragBF { v16bf v; unsigned int u[8]; };
union U4U16 { uint4 q; unsigned short e[8]; };

template <int EPI>
__global__ __launch_bounds__(256)
void gemm_bf16_kernel(const unsigned short* __restrict__ A,
                      const unsigned short* __restrict__ Bm,
                      const float* __restrict__ bias,
                      const float* __restrict__ res,
                      void* __restrict__ Out,
                      int M, int N, int K,
                      int lda, int ldb, int ldo, int HB,
                      long long sAb, long long sAh,
                      long long sBb, long long sBh,
                      long long sOb, long long sOh) {
    __shared__ unsigned short ldsA[TILE_M * LDSA_S];
    __shared__ unsigned short ldsBT[TILE_N * LDSB_S];

    int z = blockIdx.z;
    int bb = z / HB, hh = z % HB;
    const unsigned short* Ab = A + sAb * bb + sAh * hh;
    const unsigned short* Bb = Bm + sBb * bb + sBh * hh;
    long long oBase = sOb * bb + sOh * hh;

    int tid = threadIdx.x;
    int wid = tid >> 5;
    int lane = tid & 31;
    int waveM = (wid >> 1) * 32;         // 0,32,64,96
    int waveN = (wid & 1) * 64;          // 0,64
    int mTile = blockIdx.y * TILE_M;
    int nTile = blockIdx.x * TILE_N;
    int lhalf = lane >> 4;
    int l15 = lane & 15;

    // staging indices: A tile 128x32 (2 passes of 64 rows),
    //                  B tile 32x128 (2 passes of 16 rows, transposed store)
    int arow0 = tid >> 2;                // 0..63
    int acol  = (tid & 3) * 8;           // 0,8,16,24
    int brow0 = tid >> 4;                // 0..15
    int bcol  = (tid & 15) * 8;          // 0..120

    v8f acc[2][4];
#pragma unroll
    for (int i = 0; i < 2; ++i)
#pragma unroll
        for (int j = 0; j < 4; ++j) acc[i][j] = (v8f){0.f,0.f,0.f,0.f,0.f,0.f,0.f,0.f};

    for (int k0 = 0; k0 < K; k0 += TILE_K) {
        // ---- stage A (row-major) ----
#pragma unroll
        for (int p = 0; p < 2; ++p) {
            int r = arow0 + p * 64;
            int gm = mTile + r;
#if USE_ASYNC_LDS
            if (gm < M) {
                __builtin_amdgcn_global_load_async_to_lds_b128(
                    ASG(Ab + (long long)gm * lda + k0 + acol),
                    ASL(&ldsA[r * LDSA_S + acol]), 0, 0);
            } else {
                unsigned int* dst = (unsigned int*)&ldsA[r * LDSA_S + acol];
                dst[0] = 0u; dst[1] = 0u; dst[2] = 0u; dst[3] = 0u;
            }
#else
            uint4 d = make_uint4(0u, 0u, 0u, 0u);
            if (gm < M)
                d = *(const uint4*)(Ab + (long long)gm * lda + k0 + acol);
            unsigned int* dst = (unsigned int*)&ldsA[r * LDSA_S + acol];
            dst[0] = d.x; dst[1] = d.y; dst[2] = d.z; dst[3] = d.w;
#endif
        }
        // ---- stage B transposed into LDS: ldsBT[n][k] ----
#pragma unroll
        for (int p = 0; p < 2; ++p) {
            int kr = brow0 + p * 16;
            int gn = nTile + bcol;
            U4U16 tb;
            tb.q = make_uint4(0u, 0u, 0u, 0u);
            if (gn < N)
                tb.q = *(const uint4*)(Bb + (long long)(k0 + kr) * ldb + gn);
#pragma unroll
            for (int j = 0; j < 8; ++j) ldsBT[(bcol + j) * LDSB_S + kr] = tb.e[j];
        }
        // prefetch next K tile (gfx1250 global_prefetch path)
        if (k0 + TILE_K < K) {
            int gm = mTile + arow0;
            if (gm < M) __builtin_prefetch(Ab + (long long)gm * lda + k0 + TILE_K + acol, 0, 1);
            if (nTile + bcol < N)
                __builtin_prefetch(Bb + (long long)(k0 + TILE_K + brow0) * ldb + nTile + bcol, 0, 1);
        }
#if USE_ASYNC_LDS
        WAIT_ASYNC();
#endif
        __syncthreads();

        // ---- fragments (contiguous k = 16*lhalf + s, same order in A and B) ----
        FragBF fa[2], fb[4];
#pragma unroll
        for (int i = 0; i < 2; ++i) {
            int r = waveM + i * 16 + l15;
            const unsigned int* pa = (const unsigned int*)&ldsA[r * LDSA_S + lhalf * 16];
#pragma unroll
            for (int j = 0; j < 8; ++j) fa[i].u[j] = pa[j];
        }
#pragma unroll
        for (int i = 0; i < 4; ++i) {
            int c = waveN + i * 16 + l15;
            const unsigned int* pb = (const unsigned int*)&ldsBT[c * LDSB_S + lhalf * 16];
#pragma unroll
            for (int j = 0; j < 8; ++j) fb[i].u[j] = pb[j];
        }
#pragma unroll
        for (int i = 0; i < 2; ++i)
#pragma unroll
            for (int j = 0; j < 4; ++j)
                acc[i][j] = __builtin_amdgcn_wmma_f32_16x16x32_bf16(
                    false, fa[i].v, false, fb[j].v, (short)0, acc[i][j], false, false);
        __syncthreads();
    }

    // ---- epilogue: C/D layout -> lane l, vgpr r: M = r + 8*(l>>4), N = l&15 ----
#pragma unroll
    for (int i = 0; i < 2; ++i) {
#pragma unroll
        for (int j = 0; j < 4; ++j) {
            int n = nTile + waveN + j * 16 + l15;
            if (n >= N) continue;
            float bv = bias ? bias[n] : 0.0f;
#pragma unroll
            for (int r2 = 0; r2 < 8; ++r2) {
                int m = mTile + waveM + i * 16 + r2 + 8 * lhalf;
                if (m >= M) continue;
                float val = acc[i][j][r2] + bv;
                long long idx = oBase + (long long)m * ldo + n;
                if (EPI == 1) val += res[idx];
                if (EPI == 2) ((unsigned short*)Out)[idx] = f2bf(val);
                else          ((float*)Out)[idx] = val;
            }
        }
    }
}

// ---------------------------------------------------------------------------
// Host-side orchestration
// ---------------------------------------------------------------------------
static void run_gemm(int epi,
                     const unsigned short* A, const unsigned short* Bm,
                     const float* bias, const float* res, void* out,
                     int M, int N, int K, int lda, int ldb, int ldo,
                     int batch, int HB,
                     long long sAb, long long sAh, long long sBb, long long sBh,
                     long long sOb, long long sOh, hipStream_t s) {
    dim3 g((N + TILE_N - 1) / TILE_N, (M + TILE_M - 1) / TILE_M, batch);
    dim3 blk(256);
    if (epi == 0)
        gemm_bf16_kernel<0><<<g, blk, 0, s>>>(A, Bm, bias, res, out, M, N, K, lda, ldb, ldo, HB, sAb, sAh, sBb, sBh, sOb, sOh);
    else if (epi == 1)
        gemm_bf16_kernel<1><<<g, blk, 0, s>>>(A, Bm, bias, res, out, M, N, K, lda, ldb, ldo, HB, sAb, sAh, sBb, sBh, sOb, sOh);
    else
        gemm_bf16_kernel<2><<<g, blk, 0, s>>>(A, Bm, bias, res, out, M, N, K, lda, ldb, ldo, HB, sAb, sAh, sBb, sBh, sOb, sOh);
}

static void run_cvt(const float* src, unsigned short* dst, long long n, hipStream_t s) {
    long long blocks = (n + 255) / 256; if (blocks > 32768) blocks = 32768;
    f2bf_kernel<<<dim3((unsigned)blocks), 256, 0, s>>>(src, dst, n);
}

extern "C" void kernel_launch(void* const* d_in, const int* in_sizes, int n_in,
                              void* d_out, int out_size, void* d_ws, size_t ws_size,
                              hipStream_t stream) {
    (void)in_sizes; (void)n_in; (void)out_size; (void)ws_size;
    const long long Bc = 8, Tc = 2048, Dc = 1024, Nc = 512, Lc = 1024, TEc = 1024, Fc = 4096, Hc = 16;
    const long long BT = Bc * Tc;        // 16384
    const long long BN = Bc * Nc;        // 4096
    const size_t MB = 1024ull * 1024ull;

    const float* x_in  = (const float*)d_in[0];
    const float* xf_in = (const float*)d_in[1];
    const float* emb   = (const float*)d_in[2];
    const float* sa_nw = (const float*)d_in[3];  const float* sa_nb = (const float*)d_in[4];
    const float* sa_qw = (const float*)d_in[5];  const float* sa_qb = (const float*)d_in[6];
    const float* sa_kw = (const float*)d_in[7];  const float* sa_kb = (const float*)d_in[8];
    const float* sa_vw = (const float*)d_in[9];  const float* sa_vb = (const float*)d_in[10];
    const float* sa_ew = (const float*)d_in[11]; const float* sa_eb = (const float*)d_in[12];
    const float* sa_snw= (const float*)d_in[13]; const float* sa_snb= (const float*)d_in[14];
    const float* sa_ow = (const float*)d_in[15]; const float* sa_ob = (const float*)d_in[16];
    const float* ca_nw = (const float*)d_in[17]; const float* ca_nb = (const float*)d_in[18];
    const float* ca_tnw= (const float*)d_in[19]; const float* ca_tnb= (const float*)d_in[20];
    const float* ca_qw = (const float*)d_in[21]; const float* ca_qb = (const float*)d_in[22];
    const float* ca_kw = (const float*)d_in[23]; const float* ca_kb = (const float*)d_in[24];
    const float* ca_vw = (const float*)d_in[25]; const float* ca_vb = (const float*)d_in[26];
    const float* ca_ew = (const float*)d_in[27]; const float* ca_eb = (const float*)d_in[28];
    const float* ca_snw= (const float*)d_in[29]; const float* ca_snb= (const float*)d_in[30];
    const float* ca_ow = (const float*)d_in[31]; const float* ca_ob = (const float*)d_in[32];
    const float* f1w   = (const float*)d_in[33]; const float* f1b   = (const float*)d_in[34];
    const float* f2w   = (const float*)d_in[35]; const float* f2b   = (const float*)d_in[36];
    const float* f_ew  = (const float*)d_in[37]; const float* f_eb  = (const float*)d_in[38];
    const float* f_snw = (const float*)d_in[39]; const float* f_snb = (const float*)d_in[40];
    const float* f_ow  = (const float*)d_in[41]; const float* f_ob  = (const float*)d_in[42];
    const float* mask  = (const float*)d_in[43];
    float* out_f = (float*)d_out;

    char* ws = (char*)d_ws;

    // ---- bf16 weight arena ----
    unsigned short* Wb = (unsigned short*)ws;
    const long long M1 = 1048576;
    unsigned short* w_saq = Wb + 0 * M1;
    unsigned short* w_sak = Wb + 1 * M1;
    unsigned short* w_sav = Wb + 2 * M1;
    unsigned short* w_sao = Wb + 3 * M1;
    unsigned short* w_sae = Wb + 4 * M1;   // 2M
    unsigned short* w_caq = Wb + 6 * M1;
    unsigned short* w_cak = Wb + 7 * M1;
    unsigned short* w_cav = Wb + 8 * M1;
    unsigned short* w_cao = Wb + 9 * M1;
    unsigned short* w_cae = Wb + 10 * M1;  // 2M
    unsigned short* w_f1  = Wb + 12 * M1;  // 4M
    unsigned short* w_f2  = Wb + 16 * M1;  // 4M
    unsigned short* w_fe  = Wb + 20 * M1;  // 2M
    unsigned short* w_fo  = Wb + 22 * M1;  // 1M

    size_t off = 48 * MB;
    float* x1 = (float*)(ws + off); off += 64 * MB;
    float* x2 = (float*)(ws + off); off += 64 * MB;
    unsigned short* semb = (unsigned short*)(ws + off); off += 1 * MB;
    float* ef = (float*)(ws + off); off += 1 * MB;
    char* R = ws + off;

    run_cvt(sa_qw, w_saq, Dc * Dc, stream);
    run_cvt(sa_kw, w_sak, Dc * Dc, stream);
    run_cvt(sa_vw, w_sav, Dc * Dc, stream);
    run_cvt(sa_ow, w_sao, Dc * Dc, stream);
    run_cvt(sa_ew, w_sae, TEc * 2 * Dc, stream);
    run_cvt(ca_qw, w_caq, Dc * Dc, stream);
    run_cvt(ca_kw, w_cak, Lc * Dc, stream);
    run_cvt(ca_vw, w_cav, Lc * Dc, stream);
    run_cvt(ca_ow, w_cao, Dc * Dc, stream);
    run_cvt(ca_ew, w_cae, TEc * 2 * Dc, stream);
    run_cvt(f1w,  w_f1, Dc * Fc, stream);
    run_cvt(f2w,  w_f2, Fc * Dc, stream);
    run_cvt(f_ew, w_fe, TEc * 2 * Dc, stream);
    run_cvt(f_ow, w_fo, Dc * Dc, stream);

    silu_bf16_kernel<<<dim3(32), 256, 0, stream>>>(emb, semb, Bc * TEc);

    // ======================= Self-attention stage =======================
    {
        unsigned short* xn  = (unsigned short*)(R + 0);
        float* qf = (float*)(R + 32 * MB);
        float* kf = (float*)(R + 96 * MB);
        float* vf = (float*)(R + 160 * MB);
        unsigned short* qb   = (unsigned short*)(R + 224 * MB);
        unsigned short* kT   = (unsigned short*)(R + 256 * MB);
        unsigned short* vb   = (unsigned short*)(R + 288 * MB);
        unsigned short* attb = (unsigned short*)(R + 320 * MB);
        float* yf = (float*)(R + 322 * MB);
        unsigned short* sb   = (unsigned short*)(R + 386 * MB);

        ln_bf16_kernel<<<dim3((unsigned)BT), 256, 0, stream>>>(x_in, sa_nw, sa_nb, xn, (int)Dc);
        run_gemm(0, xn, w_saq, sa_qb, 0, qf, (int)BT, (int)Dc, (int)Dc, 1024, 1024, 1024, 1, 1, 0,0,0,0,0,0, stream);
        run_gemm(0, xn, w_sak, sa_kb, 0, kf, (int)BT, (int)Dc, (int)Dc, 1024, 1024, 1024, 1, 1, 0,0,0,0,0,0, stream);
        run_gemm(0, xn, w_sav, sa_vb, 0, vf, (int)BT, (int)Dc, (int)Dc, 1024, 1024, 1024, 1, 1, 0,0,0,0,0,0, stream);

        head_softmax_bf16_kernel<<<dim3((unsigned)BT), 512, 0, stream>>>(qf, qb);
        colsoftmax_T_kernel<<<dim3(4, (unsigned)Bc), 256, 0, stream>>>(kf, mask, kT, (int)Tc, (int)Dc);
        maskmul_bf16_kernel<<<dim3(32768), 256, 0, stream>>>(vf, mask, vb, BT * Dc, (int)Dc);

        run_gemm(2, kT, vb, 0, 0, attb, 64, 64, (int)Tc, (int)Tc, 1024, 64, (int)(Bc * Hc), (int)Hc,
                 Hc * 64 * Tc, 64 * Tc, Tc * Dc, 64, Hc * 4096, 4096, stream);
        run_gemm(0, qb, attb, 0, 0, yf, (int)Tc, 64, 64, 1024, 64, 1024, (int)(Bc * Hc), (int)Hc,
                 Tc * Dc, 64, Hc * 4096, 4096, Tc * Dc, 64, stream);

        run_gemm(0, semb, w_sae, sa_eb, 0, ef, (int)Bc, (int)(2 * Dc), (int)TEc, 1024, 2048, 2048, 1, 1, 0,0,0,0,0,0, stream);
        styl_silu_bf16_kernel<<<dim3((unsigned)BT), 256, 0, stream>>>(yf, ef, sa_snw, sa_snb, sb, (int)Dc, (int)Tc);
        run_gemm(1, sb, w_sao, sa_ob, x_in, x1, (int)BT, (int)Dc, (int)Dc, 1024, 1024, 1024, 1, 1, 0,0,0,0,0,0, stream);
    }

    // ======================= Cross-attention stage =======================
    {
        unsigned short* xn  = (unsigned short*)(R + 0);
        unsigned short* xfn = (unsigned short*)(R + 32 * MB);
        float* qcf = (float*)(R + 40 * MB);
        float* kcf = (float*)(R + 104 * MB);
        float* vcf = (float*)(R + 120 * MB);
        unsigned short* qcb   = (unsigned short*)(R + 136 * MB);
        unsigned short* kcT   = (unsigned short*)(R + 168 * MB);
        unsigned short* vcb   = (unsigned short*)(R + 176 * MB);
        unsigned short* attcb = (unsigned short*)(R + 184 * MB);
        float* ycf = (float*)(R + 186 * MB);
        unsigned short* scb   = (unsigned short*)(R + 250 * MB);

        ln_bf16_kernel<<<dim3((unsigned)BT), 256, 0, stream>>>(x1, ca_nw, ca_nb, xn, (int)Dc);
        ln_bf16_kernel<<<dim3((unsigned)BN), 256, 0, stream>>>(xf_in, ca_tnw, ca_tnb, xfn, (int)Lc);

        run_gemm(0, xn,  w_caq, ca_qb, 0, qcf, (int)BT, (int)Dc, (int)Dc, 1024, 1024, 1024, 1, 1, 0,0,0,0,0,0, stream);
        run_gemm(0, xfn, w_cak, ca_kb, 0, kcf, (int)BN, (int)Dc, (int)Lc, 1024, 1024, 1024, 1, 1, 0,0,0,0,0,0, stream);
        run_gemm(0, xfn, w_cav, ca_vb, 0, vcf, (int)BN, (int)Dc, (int)Lc, 1024, 1024, 1024, 1, 1, 0,0,0,0,0,0, stream);

        head_softmax_bf16_kernel<<<dim3((unsigned)BT), 512, 0, stream>>>(qcf, qcb);
        colsoftmax_T_kernel<<<dim3(4, (unsigned)Bc), 256, 0, stream>>>(kcf, (const float*)0, kcT, (int)Nc, (int)Dc);
        run_cvt(vcf, vcb, BN * Dc, stream);

        run_gemm(2, kcT, vcb, 0, 0, attcb, 64, 64, (int)Nc, (int)Nc, 1024, 64, (int)(Bc * Hc), (int)Hc,
                 Hc * 64 * Nc, 64 * Nc, Nc * Dc, 64, Hc * 4096, 4096, stream);
        run_gemm(0, qcb, attcb, 0, 0, ycf, (int)Tc, 64, 64, 1024, 64, 1024, (int)(Bc * Hc), (int)Hc,
                 Tc * Dc, 64, Hc * 4096, 4096, Tc * Dc, 64, stream);

        run_gemm(0, semb, w_cae, ca_eb, 0, ef, (int)Bc, (int)(2 * Dc), (int)TEc, 1024, 2048, 2048, 1, 1, 0,0,0,0,0,0, stream);
        styl_silu_bf16_kernel<<<dim3((unsigned)BT), 256, 0, stream>>>(ycf, ef, ca_snw, ca_snb, scb, (int)Dc, (int)Tc);
        run_gemm(1, scb, w_cao, ca_ob, x1, x2, (int)BT, (int)Dc, (int)Dc, 1024, 1024, 1024, 1, 1, 0,0,0,0,0,0, stream);
    }

    // ============================ FFN stage =============================
    {
        unsigned short* x2b = (unsigned short*)(R + 0);
        float* h1f = (float*)(R + 32 * MB);                    // 256MB
        unsigned short* h1b = (unsigned short*)(R + 288 * MB); // 128MB
        float* yff = (float*)(R + 32 * MB);                    // reuse h1f after gelu
        unsigned short* sfb = (unsigned short*)(R + 96 * MB);

        run_cvt(x2, x2b, BT * Dc, stream);
        run_gemm(0, x2b, w_f1, f1b, 0, h1f, (int)BT, (int)Fc, (int)Dc, 1024, 4096, 4096, 1, 1, 0,0,0,0,0,0, stream);
        gelu_bf16_kernel<<<dim3(32768), 256, 0, stream>>>(h1f, h1b, BT * Fc);
        run_gemm(0, h1b, w_f2, f2b, 0, yff, (int)BT, (int)Dc, (int)Fc, 4096, 1024, 1024, 1, 1, 0,0,0,0,0,0, stream);

        run_gemm(0, semb, w_fe, f_eb, 0, ef, (int)Bc, (int)(2 * Dc), (int)TEc, 1024, 2048, 2048, 1, 1, 0,0,0,0,0,0, stream);
        styl_silu_bf16_kernel<<<dim3((unsigned)BT), 256, 0, stream>>>(yff, ef, f_snw, f_snb, sfb, (int)Dc, (int)Tc);
        run_gemm(1, sfb, w_fo, f_ob, x2, out_f, (int)BT, (int)Dc, (int)Dc, 1024, 1024, 1024, 1, 1, 0,0,0,0,0,0, stream);
    }
}
